// CrossAttentionRefiner_24824910971247
// MI455X (gfx1250) — compile-verified, matrix operand order checked by reading, not run on hardware
//
#include <hip/hip_runtime.h>
#include <math.h>

// ---------------- constants ----------------
#define D_   128
#define NQ_  16384
#define TK_  128
#define B_   16
#define MLPH_ 256
#define ROWS_ 128      // query rows per block
#define STR1 136       // LDS stride (halves) for 128-col tiles (16B aligned rows)
#define STRH 264       // LDS stride (halves) for 256-col tile
// dynamic-LDS layout (halves): 3 x (128*STR1) tiles, 1 x (128*STRH), then K/V staging
#define KV_HALF_OFF (3 * ROWS_ * STR1 + ROWS_ * STRH)   // 86016 halves = 172032 bytes
#define KV_BYTE_OFF (KV_HALF_OFF * 2)

typedef __attribute__((ext_vector_type(16))) _Float16 v16h;
typedef __attribute__((ext_vector_type(8)))  float    v8f;

union FragU { v16h h; uint4 q[2]; };

#define WMMA_F16(A, B, C) \
  __builtin_amdgcn_wmma_f32_16x16x32_f16(false, (A), false, (B), (short)0, (C), false, false)

// Async copy 16B global -> LDS (tracked on ASYNCcnt). lds_off = wave LDS byte offset
// (dynamic LDS starts at 0 here: no static __shared__ in this kernel).
__device__ __forceinline__ void async_ld_b128(unsigned lds_off, const void* gptr) {
  asm volatile("global_load_async_to_lds_b128 %0, %1, off"
               :: "v"(lds_off), "v"((unsigned long long)(uintptr_t)gptr)
               : "memory");
}
__device__ __forceinline__ void wait_async0() {
  asm volatile("s_wait_asynccnt 0x0" ::: "memory");
}

// Load a 16x32 f16 fragment (A layout; B uses same pattern on transposed storage).
// p is row-major with 'stride' halves per row; r0 = tile row base, k0 = K chunk base.
__device__ __forceinline__ v16h ld_frag(const _Float16* p, int stride, int r0, int k0) {
  const int lane = threadIdx.x & 31;
  const int g    = lane >> 4;
  const _Float16* rp = p + (size_t)(r0 + (lane & 15)) * stride + k0 + g * 8;
  FragU f;
  f.q[0] = *(const uint4*)(rp);        // K = k0 + g*8 .. +7
  f.q[1] = *(const uint4*)(rp + 16);   // K = k0 + 16 + g*8 .. +7
  return f.h;
}

// One K-step of a strip GEMM: preload all NT B-fragments into distinct registers
// (2*NT b128 loads issued back-to-back in one clause), THEN issue the NT WMMAs.
// This amortizes one memory latency over NT WMMAs instead of paying it per-WMMA.
template<int NT>
__device__ __forceinline__ void gemm_step(v8f* acc, v16h a, const _Float16* W,
                                          int stride, int n0, int k0) {
  v16h bf[NT];
#pragma unroll
  for (int t = 0; t < NT; ++t) bf[t] = ld_frag(W, stride, n0 + 16 * t, k0);
#pragma unroll
  for (int t = 0; t < NT; ++t) acc[t] = WMMA_F16(a, bf[t], acc[t]);
}

// Store a 16x16 f32 C/D tile as f16 to row-major LDS (C layout: m=j+8g, n=lane%16).
__device__ __forceinline__ void st_tile(_Float16* p, int stride, int m0, int n0, const v8f acc) {
  const int lane = threadIdx.x & 31;
  const int g    = lane >> 4;
  const int n    = n0 + (lane & 15);
#pragma unroll
  for (int j = 0; j < 8; ++j)
    p[(size_t)(m0 + j + 8 * g) * stride + n] = (_Float16)acc[j];
}

// tanh-approx gelu: 1 v_exp + ~7 VALU; keeps VALU off the critical path.
__device__ __forceinline__ float gelu_f(float x) {
  float u = 0.7978845608028654f * x * (1.0f + 0.044715f * x * x);
  float e = __expf(2.0f * u);
  float th = (e - 1.0f) / (e + 1.0f);
  return 0.5f * x * (1.0f + th);
}

// ---------------- prep: weight transpose + f16 convert ----------------
// src [R=in][C=out] f32 row-major  ->  dst [C=out][R=in] f16 row-major
__global__ void wt_convert_kernel(const float* __restrict__ src, _Float16* __restrict__ dst,
                                  int R, int C) {
  int idx = blockIdx.x * blockDim.x + threadIdx.x;
  if (idx < R * C) {
    int r = idx / C, c = idx - r * C;
    dst[(size_t)c * R + r] = (_Float16)src[idx];
  }
}

// ---------------- prep: LN(T) -> K,V projections (f16) ----------------
// wK : [b][tk][128]   (scores B operand: rows = tk, cols = d)
// wVt: [b][d][128tk]  (ctx    B operand: rows = dcol, cols = tk)
__global__ void kv_prep_kernel(const float* __restrict__ T, const float* __restrict__ Wk,
                               const float* __restrict__ bk, const float* __restrict__ Wv,
                               const float* __restrict__ bv, const float* __restrict__ g_kv,
                               const float* __restrict__ b_kv,
                               _Float16* __restrict__ wK, _Float16* __restrict__ wVt) {
  __shared__ float sTn[D_];
  __shared__ float sred[8];
  const int b  = blockIdx.x >> 7;
  const int tk = blockIdx.x & 127;
  const int i  = threadIdx.x;                    // 0..127
  float x = T[((size_t)b * TK_ + tk) * D_ + i];
  float s = x, s2 = x * x;
#pragma unroll
  for (int o = 1; o < 32; o <<= 1) { s += __shfl_xor(s, o, 32); s2 += __shfl_xor(s2, o, 32); }
  if ((i & 31) == 0) { sred[i >> 5] = s; sred[4 + (i >> 5)] = s2; }
  __syncthreads();
  float ts  = sred[0] + sred[1] + sred[2] + sred[3];
  float ts2 = sred[4] + sred[5] + sred[6] + sred[7];
  float mean = ts * (1.0f / D_);
  float var  = ts2 * (1.0f / D_) - mean * mean;
  float rstd = rsqrtf(var + 1e-5f);
  sTn[i] = (x - mean) * rstd * g_kv[i] + b_kv[i];
  __syncthreads();
  float ka = bk[i], va = bv[i];
#pragma unroll 8
  for (int kk = 0; kk < D_; ++kk) {
    float t = sTn[kk];
    ka = fmaf(t, Wk[kk * D_ + i], ka);
    va = fmaf(t, Wv[kk * D_ + i], va);
  }
  wK [((size_t)b * TK_ + tk) * D_ + i]  = (_Float16)ka;
  wVt[((size_t)b * D_  + i) * TK_ + tk] = (_Float16)va;
}

// ---------------- fused main kernel ----------------
__global__ void __launch_bounds__(256, 1)
fused_attn_mlp_kernel(const float* __restrict__ Q,
    const float* __restrict__ bq,    const float* __restrict__ bo,
    const float* __restrict__ g_q,   const float* __restrict__ b_q,
    const float* __restrict__ g_out, const float* __restrict__ b_out,
    const float* __restrict__ res_scale,
    const float* __restrict__ b1,    const float* __restrict__ g_mlp,
    const float* __restrict__ b_mlp, const float* __restrict__ b2,
    const float* __restrict__ W3,    const float* __restrict__ b3,
    const _Float16* __restrict__ wWqt, const _Float16* __restrict__ wWot,
    const _Float16* __restrict__ wW1t, const _Float16* __restrict__ wW2t,
    const _Float16* __restrict__ wK,   const _Float16* __restrict__ wVt,
    float* __restrict__ dout) {
  extern __shared__ __align__(16) _Float16 smem[];
  _Float16* sQ = smem;                   // raw Q (f16)          128 x STR1
  _Float16* sA = smem + ROWS_ * STR1;    // Qn -> attn -> ctx    128 x STR1
  _Float16* sX = smem + 2 * ROWS_ * STR1;// q-proj -> attn_out   128 x STR1
  _Float16* sH = smem + 3 * ROWS_ * STR1;// gelu'd h1            128 x STRH
  _Float16* sK = smem + KV_HALF_OFF;             // K staged  [tk][128]
  _Float16* sV = smem + KV_HALF_OFF + TK_ * D_;  // Vt staged [dcol][128tk]

  const int b    = blockIdx.x >> 7;
  const int row0 = (blockIdx.x & 127) * ROWS_;
  const int lane = threadIdx.x & 31;
  const int wv   = threadIdx.x >> 5;
  const int g    = lane >> 4;
  const int l15  = lane & 15;
  const int mloc = wv * 16;              // this wave's private row strip

  // ---- Kick off async K/V -> LDS copy (overlapped with Stage 0/1) ----
  {
    const char* gk = (const char*)(wK  + (size_t)b * TK_ * D_);
    const char* gv = (const char*)(wVt + (size_t)b * D_ * TK_);
#pragma unroll
    for (int i = 0; i < 8; ++i) {
      unsigned off = (threadIdx.x + i * 256) * 16;      // 256 threads x 8 x 16B = 32KB each
      async_ld_b128(KV_BYTE_OFF + off,         gk + off);
      async_ld_b128(KV_BYTE_OFF + 32768 + off, gv + off);
    }
  }

  // ---- Stage 0: load Q tile, LayerNorm -> sA; raw f16 -> sQ (wave-private rows) ----
  {
    const int row  = threadIdx.x >> 1;           // wave w covers rows 16w..16w+15
    const int part = (threadIdx.x & 1) * 64;
    const float* qrow = Q + ((size_t)b * NQ_ + row0 + row) * D_ + part;
    float v[64];
    float s = 0.f, s2 = 0.f;
#pragma unroll
    for (int i = 0; i < 64; i += 4) {
      float4 f4 = *(const float4*)(qrow + i);
      v[i] = f4.x; v[i+1] = f4.y; v[i+2] = f4.z; v[i+3] = f4.w;
      s  += f4.x + f4.y + f4.z + f4.w;
      s2 += f4.x*f4.x + f4.y*f4.y + f4.z*f4.z + f4.w*f4.w;
    }
    s  += __shfl_xor(s, 1, 32);
    s2 += __shfl_xor(s2, 1, 32);
    float mean = s * (1.0f / D_);
    float var  = s2 * (1.0f / D_) - mean * mean;
    float rstd = rsqrtf(var + 1e-5f);
#pragma unroll 4
    for (int i = 0; i < 64; ++i) {
      int col = part + i;
      sQ[(size_t)row * STR1 + col] = (_Float16)v[i];
      sA[(size_t)row * STR1 + col] = (_Float16)((v[i] - mean) * rstd * g_q[col] + b_q[col]);
    }
  }
  // sQ/sA/sX/sH rows are wave-private -> no barriers for them anywhere.

  // ---- Stage 1: q = Qn @ Wq + bq  -> sX ----
  {
    v8f acc[8];
#pragma unroll
    for (int t = 0; t < 8; ++t) {
      float bb = bq[16 * t + l15];
#pragma unroll
      for (int j = 0; j < 8; ++j) acc[t][j] = bb;
    }
#pragma unroll
    for (int kc = 0; kc < 4; ++kc)
      gemm_step<8>(acc, ld_frag(sA, STR1, mloc, kc * 32), wWqt, D_, 0, kc * 32);
#pragma unroll
    for (int t = 0; t < 8; ++t) st_tile(sX, STR1, mloc, 16 * t, acc[t]);
  }

  // K/V staging must be complete and visible to all waves before attention.
  wait_async0();
  __syncthreads();

  // ---- Stage 2: per-head scores (one WMMA, K=HD=32), softmax, ctx, attn mean ----
  float amean[8][8];
  v8f ctx[8];
#pragma unroll
  for (int t = 0; t < 8; ++t)
#pragma unroll
    for (int j = 0; j < 8; ++j) { amean[t][j] = 0.f; ctx[t][j] = 0.f; }

  const float sscale = 0.17677669529663687f;   // 1/sqrt(32)
#pragma unroll
  for (int h = 0; h < 4; ++h) {
    v16h aq = ld_frag(sX, STR1, mloc, h * 32);
    v8f sc[8];
#pragma unroll
    for (int t = 0; t < 8; ++t)
#pragma unroll
      for (int j = 0; j < 8; ++j) sc[t][j] = 0.f;
    gemm_step<8>(sc, aq, sK, D_, 0, h * 32);
    // row softmax over Tk=128 (cols live across 8 tiles x 16 lanes of this group)
#pragma unroll
    for (int j = 0; j < 8; ++j) {
      float m = -3.0e38f;
#pragma unroll
      for (int t = 0; t < 8; ++t) { float x = sc[t][j] * sscale; sc[t][j] = x; m = fmaxf(m, x); }
#pragma unroll
      for (int o = 1; o < 16; o <<= 1) m = fmaxf(m, __shfl_xor(m, o, 32));
      float s = 0.f;
#pragma unroll
      for (int t = 0; t < 8; ++t) { float e = __expf(sc[t][j] - m); sc[t][j] = e; s += e; }
#pragma unroll
      for (int o = 1; o < 16; o <<= 1) s += __shfl_xor(s, o, 32);
      float inv = 1.0f / s;
#pragma unroll
      for (int t = 0; t < 8; ++t) { float a = sc[t][j] * inv; sc[t][j] = a; amean[t][j] += 0.25f * a; }
    }
#pragma unroll
    for (int t = 0; t < 8; ++t) st_tile(sA, STR1, mloc, 16 * t, sc[t]);   // stage attn as A
#pragma unroll
    for (int kc = 0; kc < 4; ++kc)
      gemm_step<2>(&ctx[2 * h], ld_frag(sA, STR1, mloc, kc * 32), sV, TK_, h * 32, kc * 32);
  }
  {
    float* omean = dout + (size_t)B_ * NQ_;          // residual occupies first B*Nq floats
    const size_t gr0 = (size_t)b * NQ_ + row0 + mloc + 8 * g;
#pragma unroll
    for (int j = 0; j < 8; ++j) {
      float* orow = omean + (gr0 + j) * TK_ + l15;
#pragma unroll
      for (int t = 0; t < 8; ++t) orow[16 * t] = amean[t][j];
    }
  }

  // ---- Stage 3: attn_out = ctx @ Wo + bo ----
  v8f ao[8];
  {
#pragma unroll
    for (int t = 0; t < 8; ++t) st_tile(sA, STR1, mloc, 16 * t, ctx[t]);  // stage ctx as A
#pragma unroll
    for (int t = 0; t < 8; ++t) {
      float bb = bo[16 * t + l15];
#pragma unroll
      for (int j = 0; j < 8; ++j) ao[t][j] = bb;
    }
#pragma unroll
    for (int kc = 0; kc < 4; ++kc)
      gemm_step<8>(ao, ld_frag(sA, STR1, mloc, kc * 32), wWot, D_, 0, kc * 32);
  }

  // ---- Stage 4: LN(Q + res_scale*attn_out) -> X in sX ----
  {
    const float rsc = res_scale[0];
    float gg[8], bb2[8];
#pragma unroll
    for (int t = 0; t < 8; ++t) { gg[t] = g_out[16 * t + l15]; bb2[t] = b_out[16 * t + l15]; }
#pragma unroll
    for (int t = 0; t < 8; ++t)
#pragma unroll
      for (int j = 0; j < 8; ++j) {
        float qv = (float)sQ[(size_t)(mloc + j + 8 * g) * STR1 + 16 * t + l15];
        ao[t][j] = qv + rsc * ao[t][j];
      }
#pragma unroll
    for (int j = 0; j < 8; ++j) {
      float s = 0.f, s2 = 0.f;
#pragma unroll
      for (int t = 0; t < 8; ++t) { float x = ao[t][j]; s += x; s2 += x * x; }
#pragma unroll
      for (int o = 1; o < 16; o <<= 1) { s += __shfl_xor(s, o, 32); s2 += __shfl_xor(s2, o, 32); }
      float mean = s * (1.0f / D_);
      float var  = s2 * (1.0f / D_) - mean * mean;
      float rstd = rsqrtf(var + 1e-5f);
#pragma unroll
      for (int t = 0; t < 8; ++t) ao[t][j] = (ao[t][j] - mean) * rstd * gg[t] + bb2[t];
    }
#pragma unroll
    for (int t = 0; t < 8; ++t) st_tile(sX, STR1, mloc, 16 * t, ao[t]);
  }

  // ---- Stage 5: h1 = LN(gelu([Q, X] @ W1 + b1)) -> sH ----
  // Two N-halves of 8 tiles each (64 acc VGPRs live, so B-frag groups can pipeline).
  {
    float rs_[8], rs2_[8];
#pragma unroll
    for (int j = 0; j < 8; ++j) { rs_[j] = 0.f; rs2_[j] = 0.f; }
#pragma unroll
    for (int nh = 0; nh < 2; ++nh) {
      v8f h1[8];
#pragma unroll
      for (int t = 0; t < 8; ++t) {
        float bb = b1[nh * 128 + 16 * t + l15];
#pragma unroll
        for (int j = 0; j < 8; ++j) h1[t][j] = bb;
      }
#pragma unroll
      for (int kc = 0; kc < 4; ++kc)       // K = 0..127 : Q part
        gemm_step<8>(h1, ld_frag(sQ, STR1, mloc, kc * 32), wW1t, 2 * D_, nh * 128, kc * 32);
#pragma unroll
      for (int kc = 0; kc < 4; ++kc)       // K = 128..255 : attn_out part
        gemm_step<8>(h1, ld_frag(sX, STR1, mloc, kc * 32), wW1t, 2 * D_, nh * 128, D_ + kc * 32);
#pragma unroll
      for (int t = 0; t < 8; ++t)
#pragma unroll
        for (int j = 0; j < 8; ++j) {
          float v = gelu_f(h1[t][j]);
          h1[t][j] = v; rs_[j] += v; rs2_[j] += v * v;
        }
#pragma unroll
      for (int t = 0; t < 8; ++t) st_tile(sH, STRH, mloc, nh * 128 + 16 * t, h1[t]);
    }
    // per-row mean/var over all 256 cols, then in-place LN affine (wave-private rows)
    float mean[8], rstd[8];
#pragma unroll
    for (int j = 0; j < 8; ++j) {
      float s = rs_[j], s2 = rs2_[j];
#pragma unroll
      for (int o = 1; o < 16; o <<= 1) { s += __shfl_xor(s, o, 32); s2 += __shfl_xor(s2, o, 32); }
      mean[j] = s * (1.0f / MLPH_);
      float var = s2 * (1.0f / MLPH_) - mean[j] * mean[j];
      rstd[j] = rsqrtf(var + 1e-5f);
    }
    float gm[16], bm[16];
#pragma unroll
    for (int t = 0; t < 16; ++t) { gm[t] = g_mlp[16 * t + l15]; bm[t] = b_mlp[16 * t + l15]; }
#pragma unroll
    for (int t = 0; t < 16; ++t)
#pragma unroll
      for (int j = 0; j < 8; ++j) {
        size_t idx = (size_t)(mloc + j + 8 * g) * STRH + 16 * t + l15;
        float v = (float)sH[idx];
        sH[idx] = (_Float16)((v - mean[j]) * rstd[j] * gm[t] + bm[t]);
      }
  }

  // ---- Stage 6: h2 = gelu(h1 @ W2 + b2); residual = h2 @ W3 + b3 ----
  {
    v8f h2[8];
#pragma unroll
    for (int t = 0; t < 8; ++t) {
      float bb = b2[16 * t + l15];
#pragma unroll
      for (int j = 0; j < 8; ++j) h2[t][j] = bb;
    }
#pragma unroll
    for (int kc = 0; kc < 8; ++kc)
      gemm_step<8>(h2, ld_frag(sH, STRH, mloc, kc * 32), wW2t, 2 * D_, 0, kc * 32);
    float w3v[8];
#pragma unroll
    for (int t = 0; t < 8; ++t) w3v[t] = W3[16 * t + l15];
    const float b3v = b3[0];
    const size_t gr0 = (size_t)b * NQ_ + row0 + mloc + 8 * g;
#pragma unroll
    for (int j = 0; j < 8; ++j) {
      float r = 0.f;
#pragma unroll
      for (int t = 0; t < 8; ++t) r += gelu_f(h2[t][j]) * w3v[t];
#pragma unroll
      for (int o = 1; o < 16; o <<= 1) r += __shfl_xor(r, o, 32);
      if (l15 == 0) dout[gr0 + j] = r + b3v;
    }
  }
}

extern "C" void kernel_launch(void* const* d_in, const int* in_sizes, int n_in,
                              void* d_out, int out_size, void* d_ws, size_t ws_size,
                              hipStream_t stream) {
  const float* Q        = (const float*)d_in[0];
  const float* T        = (const float*)d_in[1];
  const float* Wq       = (const float*)d_in[2];
  const float* bq       = (const float*)d_in[3];
  const float* Wk       = (const float*)d_in[4];
  const float* bk       = (const float*)d_in[5];
  const float* Wv       = (const float*)d_in[6];
  const float* bv       = (const float*)d_in[7];
  const float* Wo       = (const float*)d_in[8];
  const float* bo       = (const float*)d_in[9];
  const float* g_q      = (const float*)d_in[10];
  const float* b_q      = (const float*)d_in[11];
  const float* g_kv     = (const float*)d_in[12];
  const float* b_kv     = (const float*)d_in[13];
  const float* g_out    = (const float*)d_in[14];
  const float* b_out    = (const float*)d_in[15];
  const float* res_scale= (const float*)d_in[16];
  const float* W1       = (const float*)d_in[17];
  const float* b1       = (const float*)d_in[18];
  const float* g_mlp    = (const float*)d_in[19];
  const float* b_mlp    = (const float*)d_in[20];
  const float* W2       = (const float*)d_in[21];
  const float* b2       = (const float*)d_in[22];
  const float* W3       = (const float*)d_in[23];
  const float* b3       = (const float*)d_in[24];
  float* dout = (float*)d_out;

  // f16 workspace layout (halves)
  _Float16* wb   = (_Float16*)d_ws;
  _Float16* wWqt = wb;              // 16384
  _Float16* wWot = wb + 16384;      // 16384
  _Float16* wW1t = wb + 32768;      // 65536
  _Float16* wW2t = wb + 98304;      // 32768
  _Float16* wK   = wb + 131072;     // 262144
  _Float16* wVt  = wb + 393216;     // 262144  (total 1.25 MB)

  wt_convert_kernel<<<64,  256, 0, stream>>>(Wq, wWqt, 128, 128);
  wt_convert_kernel<<<64,  256, 0, stream>>>(Wo, wWot, 128, 128);
  wt_convert_kernel<<<256, 256, 0, stream>>>(W1, wW1t, 256, 256);
  wt_convert_kernel<<<128, 256, 0, stream>>>(W2, wW2t, 256, 128);
  kv_prep_kernel<<<2048, 128, 0, stream>>>(T, Wk, bk, Wv, bv, g_kv, b_kv, wK, wVt);

  const size_t SMEM = (size_t)KV_BYTE_OFF + 2 * TK_ * D_ * sizeof(_Float16); // 237,568 B
  (void)hipFuncSetAttribute(reinterpret_cast<const void*>(fused_attn_mlp_kernel),
                            hipFuncAttributeMaxDynamicSharedMemorySize, (int)SMEM);
  fused_attn_mlp_kernel<<<2048, 256, SMEM, stream>>>(
      Q, bq, bo, g_q, b_q, g_out, b_out, res_scale,
      b1, g_mlp, b_mlp, b2, W3, b3,
      wWqt, wWot, wW1t, wW2t, wK, wVt, dout);
}
